// MemoryEfficientCrossAttentionBlock_26671746908380
// MI455X (gfx1250) — matrix operand order repaired and checked by reading, not compiled, not run on hardware
//
#include <hip/hip_runtime.h>

// MI455X / gfx1250 implementation of MemoryEfficientCrossAttentionBlock.
// All GEMMs and both attention matmuls go through v_wmma_f32_16x16x32_f16
// (f16 inputs, f32 accumulation). Softmax / LayerNorm / GELU in f32.
// Global->LDS tile movement uses gfx1250 async-to-LDS DMA (ASYNCcnt).

typedef _Float16 half_t;
typedef __attribute__((ext_vector_type(16))) _Float16 v16h;
typedef __attribute__((ext_vector_type(8)))  _Float16 v8h;
typedef __attribute__((ext_vector_type(8)))  float    v8f;

#define LATENT 1024
#define TEXT   768
#define HEADS  16
#define HDIM   64
#define TKEY   256
#define FF     4096
#define ATTN_SCALE 0.125f       // HDIM^-0.5
#define LN_EPS 1e-5f

__device__ __forceinline__ v16h cat16(v8h lo, v8h hi) {
  v16h r;
#pragma unroll
  for (int i = 0; i < 8; i++) { r[i] = lo[i]; r[i + 8] = hi[i]; }
  return r;
}

__device__ __forceinline__ v8f wmma16(v16h a, v16h b, v8f c) {
  // (neg_a, A, neg_b, B, c_mod, C, reuse_a, reuse_b)
  return __builtin_amdgcn_wmma_f32_16x16x32_f16(false, a, false, b, (short)0, c,
                                                false, false);
}

// LDS offset of a shared-memory pointer (shared aperture keeps the AS3 offset
// in the low 32 bits of the flat address).
__device__ __forceinline__ uint32_t lds_off(const void* p) {
  return (uint32_t)(uintptr_t)p;
}

// gfx1250 async DMA: copy 16B per lane from global to LDS; INST_OFFSET applies
// to both the global and the LDS address. Tracked with ASYNCcnt.
template <int OFF>
__device__ __forceinline__ void async_b128(uint32_t lds, const void* g) {
  asm volatile("global_load_async_to_lds_b128 %0, %1, off offset:%2"
               :: "v"(lds), "v"(g), "n"(OFF) : "memory");
}

// 32 bytes per lane (one 16-half row segment): two async ops.
__device__ __forceinline__ void async_row32(uint32_t lds, const void* g) {
  async_b128<0>(lds, g);
  async_b128<16>(lds, g);
}

// ---------------------------------------------------------------------------
// f32 -> f16 elementwise cast (4 elems / thread)
// ---------------------------------------------------------------------------
__global__ __launch_bounds__(256) void cast_f32_f16(const float* __restrict__ x,
                                                    half_t* __restrict__ y, int n4) {
  int i = blockIdx.x * blockDim.x + threadIdx.x;
  if (i < n4) {
    const float4 v = *(const float4*)(x + (size_t)i * 4);
    half_t* o = y + (size_t)i * 4;
    o[0] = (half_t)v.x; o[1] = (half_t)v.y; o[2] = (half_t)v.z; o[3] = (half_t)v.w;
  }
}

// ---------------------------------------------------------------------------
// Weight transpose + cast:  W f32 [Kd][Nd]  ->  Wt f16 [Nd][Kd]
// ---------------------------------------------------------------------------
__global__ __launch_bounds__(256) void transpose_cast_w(const float* __restrict__ W,
                                                        half_t* __restrict__ Wt,
                                                        int Kd, int Nd) {
  __shared__ float t[32][33];
  const int n0 = blockIdx.x * 32, k0 = blockIdx.y * 32;
  const int tx = threadIdx.x, ty = threadIdx.y; // 32 x 8
#pragma unroll
  for (int i = 0; i < 32; i += 8)
    t[ty + i][tx] = W[(size_t)(k0 + ty + i) * Nd + n0 + tx];
  __syncthreads();
#pragma unroll
  for (int i = 0; i < 32; i += 8)
    Wt[(size_t)(n0 + ty + i) * Kd + k0 + tx] = (half_t)t[tx][ty + i];
}

// ---------------------------------------------------------------------------
// Generic WMMA GEMM:  C[M][N] = A[M][K] (f16, row-major) * Wt[N][K] (f16) + bias
// Block tile 128x128, BK=32, 8 waves (2x4), each wave: 4x2 16x16 frags.
// Double-buffered LDS fed by async-to-LDS DMA; ASYNCcnt pipelining.
// EPI: 0 = f16 out, 1 = f16 out + exact GELU, 2 = f32 out,
//      3 = f32 out + res1 + res2 (final  x + out + ffn)
// ---------------------------------------------------------------------------
template <int EPI>
__global__ __launch_bounds__(256) void gemm128(
    const half_t* __restrict__ A, const half_t* __restrict__ Bt,
    const float* __restrict__ bias,
    half_t* __restrict__ Ch, float* __restrict__ Cf,
    const float* __restrict__ res1, const float* __restrict__ res2,
    int M, int N, int K) {
  __shared__ half_t As[2][128 * 32];
  __shared__ half_t Bs[2][128 * 32];
  (void)M;
  const int tid = threadIdx.x;
  const int lane = tid & 31;
  const int wave = tid >> 5;
  const int wm = wave >> 2;            // 0..1 -> 64-row slab
  const int wn = wave & 3;             // 0..3 -> 32-col slab
  const int bm0 = blockIdx.y * 128;
  const int bn0 = blockIdx.x * 128;

  const int lrow = tid >> 1;           // 0..127
  const int lcol = (tid & 1) * 16;     // 0 or 16 halves

  v8f acc[4][2] = {};

  const int l15 = lane & 15;
  const int c0a = (lane & 16) ? 8 : 0;   // A-frag K base within 32-chunk
  const int c0b = (lane & 16) ? 8 : 0;   // B-frag K base within 32-chunk

  // per-thread DMA source / destination bases
  const half_t* gA = A + (size_t)(bm0 + lrow) * K + lcol;
  const half_t* gB = Bt + (size_t)(bn0 + lrow) * K + lcol;
  const uint32_t ldsA = lds_off(&As[0][lrow * 32 + lcol]);
  const uint32_t ldsB = lds_off(&Bs[0][lrow * 32 + lcol]);
  const uint32_t bufstride = 128 * 32 * 2; // bytes between double buffers

  // prologue: tile 0 -> buffer 0 (4 async ops)
  async_row32(ldsA, gA);
  async_row32(ldsB, gB);

  const int nsteps = K >> 5;
  for (int s = 0; s < nsteps; s++) {
    const int buf = s & 1;
    if (s + 1 < nsteps) {
      // prefetch tile s+1 into the other buffer, then retire tile s
      // (async loads complete in order -> wait for all but the newest 4)
      const uint32_t bo = (uint32_t)(buf ^ 1) * bufstride;
      async_row32(ldsA + bo, gA + (size_t)(s + 1) * 32);
      async_row32(ldsB + bo, gB + (size_t)(s + 1) * 32);
      asm volatile("s_wait_asynccnt 0x4" ::: "memory");
    } else {
      asm volatile("s_wait_asynccnt 0x0" ::: "memory");
    }
    __syncthreads();

    v16h af[4], bf[2];
#pragma unroll
    for (int i = 0; i < 4; i++) {
      const half_t* p = As[buf] + (wm * 64 + i * 16 + l15) * 32;
      af[i] = cat16(*(const v8h*)(p + c0a), *(const v8h*)(p + c0a + 16));
    }
#pragma unroll
    for (int j = 0; j < 2; j++) {
      const half_t* p = Bs[buf] + (wn * 32 + j * 16 + l15) * 32;
      bf[j] = cat16(*(const v8h*)(p + c0b), *(const v8h*)(p + c0b + 16));
    }
#pragma unroll
    for (int i = 0; i < 4; i++)
#pragma unroll
      for (int j = 0; j < 2; j++) acc[i][j] = wmma16(af[i], bf[j], acc[i][j]);
    __syncthreads(); // protect buf before it is DMA-overwritten by the next step's prefetch
  }

  const int hi8 = (lane & 16) ? 8 : 0;
#pragma unroll
  for (int i = 0; i < 4; i++) {
    const int grow0 = bm0 + wm * 64 + i * 16 + hi8;
#pragma unroll
    for (int j = 0; j < 2; j++) {
      const int gcol = bn0 + wn * 32 + j * 16 + l15;
      const float bv = bias ? bias[gcol] : 0.f;
#pragma unroll
      for (int r = 0; r < 8; r++) {
        float v = acc[i][j][r] + bv;
        const size_t idx = (size_t)(grow0 + r) * N + gcol;
        if (EPI == 0) {
          Ch[idx] = (half_t)v;
        } else if (EPI == 1) {
          v = 0.5f * v * (1.f + erff(v * 0.70710678118f)); // exact GELU
          Ch[idx] = (half_t)v;
        } else if (EPI == 2) {
          Cf[idx] = v;
        } else {
          Cf[idx] = v + res1[idx] + res2[idx];
        }
      }
    }
  }
}

// ---------------------------------------------------------------------------
// Fused cross-attention: one block = one (b,h) and 128 queries.
// All 256 keys fit in LDS; scores & P.V via WMMA; softmax in registers.
// ---------------------------------------------------------------------------
__global__ __launch_bounds__(256) void attn_fused(
    const half_t* __restrict__ Qh, const half_t* __restrict__ Kh,
    const half_t* __restrict__ Vh, half_t* __restrict__ Oh, int Tx) {
  __shared__ half_t KV[TKEY * HDIM];       // phase 1: K[256][64]; phase 2: Vt[64][256]
  __shared__ half_t Pbuf[8 * 16 * 32];     // per-wave P staging (D-layout -> A-layout)

  const int tid = threadIdx.x, lane = tid & 31, wave = tid >> 5;
  const int bh = blockIdx.x;
  const int b = bh >> 4, h = bh & 15;
  const int q0 = blockIdx.y * 128;
  const size_t kvrow = (size_t)(b * TKEY + tid) * LATENT + h * HDIM;

  // ---- phase 1: K rows DMA'd into LDS (B-fragment friendly: Bt[t][d]) ----
  {
    const uint32_t l = lds_off(KV + tid * HDIM);
    const half_t* g = Kh + kvrow;
    async_b128<0>(l, g);   async_b128<16>(l, g);
    async_b128<32>(l, g);  async_b128<48>(l, g);
    async_b128<64>(l, g);  async_b128<80>(l, g);
    async_b128<96>(l, g);  async_b128<112>(l, g);
    asm volatile("s_wait_asynccnt 0x0" ::: "memory");
  }
  __syncthreads();

  const int l15 = lane & 15;
  const int c0a = (lane & 16) ? 8 : 0;
  const int c0b = (lane & 16) ? 8 : 0;

  // Q A-fragments straight from global (read once)
  const half_t* qrow = Qh + (size_t)(b * Tx + q0 + wave * 16 + l15) * LATENT + h * HDIM;
  const v16h qa0 = cat16(*(const v8h*)(qrow + c0a), *(const v8h*)(qrow + c0a + 16));
  const v16h qa1 = cat16(*(const v8h*)(qrow + 32 + c0a), *(const v8h*)(qrow + 32 + c0a + 16));

  // ---- scores: 16 query rows x 256 keys, 32 WMMAs per wave ----
  v8f sc[16];
#pragma unroll
  for (int tt = 0; tt < 16; tt++) {
    const half_t* kp = KV + (tt * 16 + l15) * HDIM;
    v16h kb0 = cat16(*(const v8h*)(kp + c0b), *(const v8h*)(kp + c0b + 16));
    v16h kb1 = cat16(*(const v8h*)(kp + 32 + c0b), *(const v8h*)(kp + 32 + c0b + 16));
    v8f a = {};
    a = wmma16(qa0, kb0, a);
    a = wmma16(qa1, kb1, a);
    sc[tt] = a;
  }
  __syncthreads(); // everyone done reading K from LDS

  // ---- phase 2: Vt[64][256] overwrites the K region ----
  {
    const half_t* s = Vh + kvrow;
#pragma unroll
    for (int d = 0; d < HDIM; d++) KV[d * TKEY + tid] = s[d];
  }

  // ---- softmax in registers (half-wave xor reductions match D layout) ----
  float rmax[8], rsum[8], rinv[8];
#pragma unroll
  for (int r = 0; r < 8; r++) rmax[r] = -3.0e38f;
#pragma unroll
  for (int tt = 0; tt < 16; tt++)
#pragma unroll
    for (int r = 0; r < 8; r++) rmax[r] = fmaxf(rmax[r], sc[tt][r]);
#pragma unroll
  for (int m = 1; m < 16; m <<= 1)
#pragma unroll
    for (int r = 0; r < 8; r++) rmax[r] = fmaxf(rmax[r], __shfl_xor(rmax[r], m, 32));
#pragma unroll
  for (int r = 0; r < 8; r++) rsum[r] = 0.f;
#pragma unroll
  for (int tt = 0; tt < 16; tt++)
#pragma unroll
    for (int r = 0; r < 8; r++) {
      float p = __expf((sc[tt][r] - rmax[r]) * ATTN_SCALE);
      sc[tt][r] = p;
      rsum[r] += p;
    }
#pragma unroll
  for (int m = 1; m < 16; m <<= 1)
#pragma unroll
    for (int r = 0; r < 8; r++) rsum[r] += __shfl_xor(rsum[r], m, 32);
#pragma unroll
  for (int r = 0; r < 8; r++) rinv[r] = 1.f / rsum[r];

  __syncthreads(); // Vt fully in LDS

  // ---- P @ V : stage P chunks through per-wave LDS, 32 WMMAs per wave ----
  half_t* Pw = Pbuf + wave * 16 * 32;
  const int hi8 = (lane & 16) ? 8 : 0;
  v8f oacc[4] = {};
#pragma unroll
  for (int kk = 0; kk < 8; kk++) {
#pragma unroll
    for (int hh = 0; hh < 2; hh++) {
      const int tt = kk * 2 + hh;
#pragma unroll
      for (int r = 0; r < 8; r++)
        Pw[(r + hi8) * 32 + hh * 16 + l15] = (half_t)(sc[tt][r] * rinv[r]);
    }
    asm volatile("s_wait_dscnt 0" ::: "memory"); // per-wave LDS RAW fence
    const half_t* pp = Pw + l15 * 32;
    const v16h pa = cat16(*(const v8h*)(pp + c0a), *(const v8h*)(pp + c0a + 16));
#pragma unroll
    for (int nf = 0; nf < 4; nf++) {
      const half_t* vp = KV + (nf * 16 + l15) * TKEY + kk * 32;
      v16h vb = cat16(*(const v8h*)(vp + c0b), *(const v8h*)(vp + c0b + 16));
      oacc[nf] = wmma16(pa, vb, oacc[nf]);
    }
  }

  // ---- store attn output (f16, [B*Tx][LATENT], head columns) ----
  const size_t orow0 = (size_t)(b * Tx + q0 + wave * 16 + hi8);
#pragma unroll
  for (int nf = 0; nf < 4; nf++) {
    const int col = h * HDIM + nf * 16 + l15;
#pragma unroll
    for (int r = 0; r < 8; r++)
      Oh[(orow0 + r) * LATENT + col] = (half_t)oacc[nf][r];
  }
}

// ---------------------------------------------------------------------------
// Row LayerNorm: f32 [rows][1024] -> f16 normalized
// ---------------------------------------------------------------------------
__global__ __launch_bounds__(256) void layernorm_rows(const float* __restrict__ X,
                                                      const float* __restrict__ g,
                                                      const float* __restrict__ bb,
                                                      half_t* __restrict__ Y) {
  const int row = blockIdx.x, tid = threadIdx.x;
  const float4 v = *(const float4*)(X + (size_t)row * LATENT + tid * 4);
  float s = v.x + v.y + v.z + v.w;
  float s2 = v.x * v.x + v.y * v.y + v.z * v.z + v.w * v.w;
#pragma unroll
  for (int m = 1; m < 32; m <<= 1) { s += __shfl_xor(s, m, 32); s2 += __shfl_xor(s2, m, 32); }
  __shared__ float rs[8], rq[8];
  if ((tid & 31) == 0) { rs[tid >> 5] = s; rq[tid >> 5] = s2; }
  __syncthreads();
  float ts = 0.f, tq = 0.f;
#pragma unroll
  for (int i = 0; i < 8; i++) { ts += rs[i]; tq += rq[i]; }
  const float mu = ts * (1.f / LATENT);
  const float var = tq * (1.f / LATENT) - mu * mu;
  const float ri = rsqrtf(var + LN_EPS);
  half_t* y = Y + (size_t)row * LATENT + tid * 4;
  const float* gp = g + tid * 4;
  const float* bp = bb + tid * 4;
  y[0] = (half_t)((v.x - mu) * ri * gp[0] + bp[0]);
  y[1] = (half_t)((v.y - mu) * ri * gp[1] + bp[1]);
  y[2] = (half_t)((v.z - mu) * ri * gp[2] + bp[2]);
  y[3] = (half_t)((v.w - mu) * ri * gp[3] + bp[3]);
}

// ---------------------------------------------------------------------------
extern "C" void kernel_launch(void* const* d_in, const int* in_sizes, int n_in,
                              void* d_out, int out_size, void* d_ws, size_t ws_size,
                              hipStream_t stream) {
  (void)in_sizes; (void)n_in; (void)out_size; (void)ws_size;
  const float* x   = (const float*)d_in[0];
  const float* xf  = (const float*)d_in[1];
  const float* Wq  = (const float*)d_in[2];
  const float* bq  = (const float*)d_in[3];
  const float* Wk  = (const float*)d_in[4];
  const float* bk  = (const float*)d_in[5];
  const float* Wv  = (const float*)d_in[6];
  const float* bv  = (const float*)d_in[7];
  const float* Wo  = (const float*)d_in[8];
  const float* bo  = (const float*)d_in[9];
  const float* lng = (const float*)d_in[10];
  const float* lnb = (const float*)d_in[11];
  const float* W1  = (const float*)d_in[12];
  const float* b1  = (const float*)d_in[13];
  const float* W2  = (const float*)d_in[14];
  const float* b2  = (const float*)d_in[15];
  float* out = (float*)d_out;

  const int B = 4, Tx = 4096;
  const size_t NTOK = (size_t)B * Tx;   // 16384 query tokens
  const size_t NKV  = (size_t)B * TKEY; // 1024 key/value tokens

  char* ws = (char*)d_ws;
  size_t off = 0;
  auto take = [&](size_t bytes) -> char* {
    char* p = ws + off;
    off += (bytes + 255) & ~(size_t)255;
    return p;
  };
  half_t* WqT = (half_t*)take((size_t)LATENT * LATENT * 2);
  half_t* WkT = (half_t*)take((size_t)LATENT * TEXT * 2);
  half_t* WvT = (half_t*)take((size_t)LATENT * TEXT * 2);
  half_t* WoT = (half_t*)take((size_t)LATENT * LATENT * 2);
  half_t* W1T = (half_t*)take((size_t)FF * LATENT * 2);
  half_t* W2T = (half_t*)take((size_t)LATENT * FF * 2);
  half_t* xh   = (half_t*)take(NTOK * LATENT * 2);
  half_t* xfh  = (half_t*)take(NKV * TEXT * 2);
  half_t* Qh   = (half_t*)take(NTOK * LATENT * 2);
  half_t* Khb  = (half_t*)take(NKV * LATENT * 2);
  half_t* Vhb  = (half_t*)take(NKV * LATENT * 2);
  half_t* Ah   = (half_t*)take(NTOK * LATENT * 2);
  float*  outf = (float*) take(NTOK * LATENT * 4);
  half_t* hln  = (half_t*)take(NTOK * LATENT * 2);
  half_t* h1   = (half_t*)take(NTOK * FF * 2);

  // weights: transpose + cast to f16 [N][K]
  dim3 tb(32, 8);
  transpose_cast_w<<<dim3(LATENT / 32, LATENT / 32), tb, 0, stream>>>(Wq, WqT, LATENT, LATENT);
  transpose_cast_w<<<dim3(LATENT / 32, TEXT / 32),   tb, 0, stream>>>(Wk, WkT, TEXT, LATENT);
  transpose_cast_w<<<dim3(LATENT / 32, TEXT / 32),   tb, 0, stream>>>(Wv, WvT, TEXT, LATENT);
  transpose_cast_w<<<dim3(LATENT / 32, LATENT / 32), tb, 0, stream>>>(Wo, WoT, LATENT, LATENT);
  transpose_cast_w<<<dim3(FF / 32, LATENT / 32),     tb, 0, stream>>>(W1, W1T, LATENT, FF);
  transpose_cast_w<<<dim3(LATENT / 32, FF / 32),     tb, 0, stream>>>(W2, W2T, FF, LATENT);

  // activations to f16
  cast_f32_f16<<<(unsigned)(NTOK * LATENT / 4 + 255) / 256, 256, 0, stream>>>(
      x, xh, (int)(NTOK * LATENT / 4));
  cast_f32_f16<<<(unsigned)(NKV * TEXT / 4 + 255) / 256, 256, 0, stream>>>(
      xf, xfh, (int)(NKV * TEXT / 4));

  // Q/K/V projections
  gemm128<0><<<dim3(LATENT / 128, (unsigned)(NTOK / 128)), 256, 0, stream>>>(
      xh, WqT, bq, Qh, nullptr, nullptr, nullptr, (int)NTOK, LATENT, LATENT);
  gemm128<0><<<dim3(LATENT / 128, (unsigned)(NKV / 128)), 256, 0, stream>>>(
      xfh, WkT, bk, Khb, nullptr, nullptr, nullptr, (int)NKV, LATENT, TEXT);
  gemm128<0><<<dim3(LATENT / 128, (unsigned)(NKV / 128)), 256, 0, stream>>>(
      xfh, WvT, bv, Vhb, nullptr, nullptr, nullptr, (int)NKV, LATENT, TEXT);

  // fused attention
  attn_fused<<<dim3(B * HEADS, Tx / 128), 256, 0, stream>>>(Qh, Khb, Vhb, Ah, Tx);

  // output projection (f32, kept for LN + residual)
  gemm128<2><<<dim3(LATENT / 128, (unsigned)(NTOK / 128)), 256, 0, stream>>>(
      Ah, WoT, bo, nullptr, outf, nullptr, nullptr, (int)NTOK, LATENT, LATENT);

  // LayerNorm -> f16
  layernorm_rows<<<(unsigned)NTOK, 256, 0, stream>>>(outf, lng, lnb, hln);

  // FFN up + exact GELU
  gemm128<1><<<dim3(FF / 128, (unsigned)(NTOK / 128)), 256, 0, stream>>>(
      hln, W1T, b1, h1, nullptr, nullptr, nullptr, (int)NTOK, FF, LATENT);

  // FFN down + both residuals: out = x + attn_proj + ffn
  gemm128<3><<<dim3(LATENT / 128, (unsigned)(NTOK / 128)), 256, 0, stream>>>(
      h1, W2T, b2, nullptr, out, outf, x, (int)NTOK, LATENT, FF);
}
